// Performers_86328842650319
// MI455X (gfx1250) — matrix operand order repaired and checked by reading, not compiled, not run on hardware
//
#include <hip/hip_runtime.h>

// Performer (FAVOR+) causal attention for MI455X (gfx1250, wave32, WMMA).
// B=4 L=4096 E=512 H=8 M=128 DH=64. Memory-bound (~300MB vs 43 GFLOP), all
// GEMMs mapped to v_wmma_f32_16x16x32_bf16 (f32 accumulate); the causal scan
// is rewritten as chunked linear attention (chunk=128) so it is GEMM work too.

typedef __attribute__((ext_vector_type(16))) __bf16 v16bf;
typedef __attribute__((ext_vector_type(8)))  float  v8f;
typedef __attribute__((ext_vector_type(4)))  int    v4i;

#define AS1 __attribute__((address_space(1)))
#define AS3 __attribute__((address_space(3)))

#if __has_builtin(__builtin_amdgcn_global_load_async_to_lds_b128)
#define USE_ASYNC_KV 1
__device__ inline void wait_async0() {
#if __has_builtin(__builtin_amdgcn_s_wait_asynccnt)
  __builtin_amdgcn_s_wait_asynccnt(0);
#else
  asm volatile("s_wait_asynccnt 0x0" ::: "memory");
#endif
}
#else
#define USE_ASYNC_KV 0
#endif

static constexpr int   Bdim = 4, Ldim = 4096, Edim = 512, Hdim = 8, Mdim = 128, DHdim = 64;
static constexpr float DN    = 0.35355339059327379f;   // DH^-0.25
static constexpr float RATIO = 0.08838834764831845f;   // M^-0.5
static constexpr float EPS   = 1e-5f;

__device__ inline v8f v8f_zero() {
  v8f z;
#pragma unroll
  for (int i = 0; i < 8; ++i) z[i] = 0.f;
  return z;
}

__device__ inline v8f wmma_bf16(v16bf a, v16bf b, v8f c) {
  // D = A(16x32 bf16) x B(32x16 bf16) + C(16x16 f32)
  return __builtin_amdgcn_wmma_f32_16x16x32_bf16(false, a, false, b, (short)0, c, false, false);
}

// ---- fragment loaders (ISA 7.12.2 VGPR layouts, wave32) --------------------
// A row-major [16+ rows][lda], element A[i][k]; lane<16: row=lane, K k0..k0+7 &
// k0+16..23 ; lane>=16: row=lane-16, K k0+8..15 & k0+24..31.
__device__ inline v16bf load_a_f32(const float* __restrict__ A, int lda, int k0) {
  const int lane = threadIdx.x & 31, half = lane >> 4, r = lane & 15;
  const float* p = A + r * lda + k0 + half * 8;
  v16bf a;
#pragma unroll
  for (int j = 0; j < 8; ++j) { a[j] = (__bf16)p[j]; a[8 + j] = (__bf16)p[16 + j]; }
  return a;
}
__device__ inline v16bf load_a_bf16(const __bf16* A, int lda, int k0) {
  const int lane = threadIdx.x & 31, half = lane >> 4, r = lane & 15;
  const __bf16* p = A + r * lda + k0 + half * 8;
  v16bf a;
#pragma unroll
  for (int j = 0; j < 8; ++j) { a[j] = p[j]; a[8 + j] = p[16 + j]; }
  return a;
}
// B given N-major (Bt[n][k]); lane<16: col=lane, K k0..k0+7 & k0+16..23; lane>=16: K k0+8..15 & k0+24..31.
__device__ inline v16bf load_b_f32(const float* __restrict__ Bt, int ldb, int k0) {
  const int lane = threadIdx.x & 31, half = lane >> 4, r = lane & 15;
  const float* p = Bt + r * ldb + k0 + half * 8;
  v16bf b;
#pragma unroll
  for (int j = 0; j < 8; ++j) { b[j] = (__bf16)p[j]; b[8 + j] = (__bf16)p[16 + j]; }
  return b;
}
__device__ inline v16bf load_b_bf16(const __bf16* Bt, int ldb, int k0) {
  const int lane = threadIdx.x & 31, half = lane >> 4, r = lane & 15;
  const __bf16* p = Bt + r * ldb + k0 + half * 8;
  v16bf b;
#pragma unroll
  for (int j = 0; j < 8; ++j) { b[j] = p[j]; b[8 + j] = p[16 + j]; }
  return b;
}

// ordered-uint encoding of float for atomicMax
__device__ inline unsigned f2u(float f) {
  unsigned x = __float_as_uint(f);
  return (x & 0x80000000u) ? ~x : (x | 0x80000000u);
}
__device__ inline float u2f(unsigned u) {
  unsigned x = (u & 0x80000000u) ? (u & 0x7fffffffu) : ~u;
  return __uint_as_float(x);
}

// ---- Y[MR,N] = X[MR,K] @ W[N,K]^T + bias ----------------------------------
// Two interleaved accumulators break the WMMA->WMMA RAW chain (bf16 WMMA has a
// ~5-slot hazard, ISA 7.12.1), letting loads and wmma of the two chains overlap.
__global__ __launch_bounds__(256)
void gemm_xwt_bias(const float* __restrict__ X, const float* __restrict__ W,
                   const float* __restrict__ bias, float* __restrict__ Y,
                   int K, int N) {
  const int wave = threadIdx.x >> 5, lane = threadIdx.x & 31;
  const int half = lane >> 4, r = lane & 15;
  const long row0 = (long)blockIdx.x * 16;
  const int  col0 = (blockIdx.y * 8 + wave) * 16;
  const float* Arow = X + row0 * K;
  const float* Brow = W + (long)col0 * K;  // W is row-major [N][K] == Bt
  v8f c0 = v8f_zero(), c1 = v8f_zero();
  for (int k0 = 0; k0 < K; k0 += 64) {
    c0 = wmma_bf16(load_a_f32(Arow, K, k0),      load_b_f32(Brow, K, k0),      c0);
    c1 = wmma_bf16(load_a_f32(Arow, K, k0 + 32), load_b_f32(Brow, K, k0 + 32), c1);
  }
#pragma unroll
  for (int v = 0; v < 8; ++v) {
    long row = row0 + half * 8 + v;
    int  col = col0 + r;
    Y[row * N + col] = c0[v] + c1[v] + bias[col];
  }
}

__global__ void init_bhmax(unsigned* p) { if (threadIdx.x < 32) p[threadIdx.x] = 0u; }

// ---- FAVOR feature map: dash = dn*(D @ proj^T), diag, row/global max -------
// D: [R=B*L*H, 64]; proj: [128, 64]; out: [R, 128]
__global__ __launch_bounds__(256)
void favor_features(const float* __restrict__ D, const float* __restrict__ proj,
                    float* __restrict__ out, unsigned* __restrict__ bhmax, int isQuery) {
  __shared__ float diag_s[8][16];
  const int wave = threadIdx.x >> 5, lane = threadIdx.x & 31;
  const int half = lane >> 4, r = lane & 15;
  const long row0 = ((long)blockIdx.x * 8 + wave) * 16;
  if (lane < 16) {
    const float* p = D + (row0 + lane) * DHdim;
    float s = 0.f;
#pragma unroll
    for (int d = 0; d < DHdim; ++d) s += p[d] * p[d];
    diag_s[wave][lane] = 0.5f * DN * DN * s;
  }
  v8f acc[8];
#pragma unroll
  for (int t = 0; t < 8; ++t) acc[t] = v8f_zero();
  const float* Arow = D + row0 * DHdim;
  for (int k0 = 0; k0 < DHdim; k0 += 32) {
    v16bf a = load_a_f32(Arow, DHdim, k0);
#pragma unroll
    for (int t = 0; t < 8; ++t)
      acc[t] = wmma_bf16(a, load_b_f32(proj + (long)t * 16 * DHdim, DHdim, k0), acc[t]);
  }
  float rm[8];
#pragma unroll
  for (int v = 0; v < 8; ++v) rm[v] = -3.0e38f;
#pragma unroll
  for (int t = 0; t < 8; ++t)
#pragma unroll
    for (int v = 0; v < 8; ++v) { acc[t][v] *= DN; rm[v] = fmaxf(rm[v], acc[t][v]); }
#pragma unroll
  for (int v = 0; v < 8; ++v)
#pragma unroll
    for (int off = 1; off < 16; off <<= 1)        // reduce within 16-lane half
      rm[v] = fmaxf(rm[v], __shfl_xor(rm[v], off, 32));
  __syncthreads();
  if (isQuery) {
#pragma unroll
    for (int t = 0; t < 8; ++t)
#pragma unroll
      for (int v = 0; v < 8; ++v) {
        int rl = half * 8 + v;
        out[(row0 + rl) * Mdim + t * 16 + r] =
            RATIO * (__expf(acc[t][v] - diag_s[wave][rl] - rm[v]) + EPS);
      }
  } else {
#pragma unroll
    for (int t = 0; t < 8; ++t)
#pragma unroll
      for (int v = 0; v < 8; ++v) {
        int rl = half * 8 + v;
        out[(row0 + rl) * Mdim + t * 16 + r] = acc[t][v] - diag_s[wave][rl];
      }
    if (r == 0) {                  // h of local row (half*8+v) is v since row0%16==0
      long b = row0 >> 15;         // / (L*H)
#pragma unroll
      for (int v = 0; v < 8; ++v) atomicMax(&bhmax[b * 8 + v], f2u(rm[v]));
    }
  }
}

__global__ void exp_keys(float* __restrict__ kp, const unsigned* __restrict__ bhmax, long n) {
  long i = (long)blockIdx.x * blockDim.x + threadIdx.x;
  long stride = (long)gridDim.x * blockDim.x;
  for (; i < n; i += stride) {
    long row = i >> 7;
    int  h = (int)(row & 7);
    long b = row >> 15;
    kp[i] = RATIO * (__expf(kp[i] - u2f(bhmax[b * 8 + h])) + EPS);
  }
}

// ---- per-chunk KV = Kp^T @ V (128x64) and ksum -----------------------------
__global__ __launch_bounds__(256)
void chunk_kv(const float* __restrict__ kp, const float* __restrict__ vv,
              float* __restrict__ KVc, float* __restrict__ ksc) {
  __shared__ __bf16 KpT[128][132];
  __shared__ __bf16 VT[64][136];
  const int chunk = blockIdx.x, bh = blockIdx.y;
  const int b = bh >> 3, h = bh & 7;
  const long rowBase = ((long)b * Ldim + chunk * 128) * Hdim + h;
  for (int idx = threadIdx.x; idx < 128 * 128; idx += 256) {
    int l = idx >> 7, m = idx & 127;
    KpT[m][l] = (__bf16)kp[(rowBase + (long)l * Hdim) * Mdim + m];
  }
  for (int idx = threadIdx.x; idx < 128 * 64; idx += 256) {
    int l = idx >> 6, d = idx & 63;
    VT[d][l] = (__bf16)vv[(rowBase + (long)l * Hdim) * DHdim + d];
  }
  __syncthreads();
  if (threadIdx.x < 128) {
    float s = 0.f;
#pragma unroll
    for (int l = 0; l < 128; ++l) s += (float)KpT[threadIdx.x][l];
    ksc[((long)bh * 32 + chunk) * 128 + threadIdx.x] = s;
  }
  const int wave = threadIdx.x >> 5, lane = threadIdx.x & 31;
  const int half = lane >> 4, r = lane & 15;
  const int m0 = wave * 16;
  float* outp = KVc + ((long)bh * 32 + chunk) * 128 * 64;
  for (int dt = 0; dt < 4; ++dt) {
    v8f c = v8f_zero();
    for (int k0 = 0; k0 < 128; k0 += 32)
      c = wmma_bf16(load_a_bf16(&KpT[m0][0], 132, k0), load_b_bf16(&VT[dt * 16][0], 136, k0), c);
#pragma unroll
    for (int v = 0; v < 8; ++v)
      outp[(m0 + half * 8 + v) * 64 + dt * 16 + r] = c[v];
  }
}

// ---- exclusive prefix over 32 chunks per (b,h) -----------------------------
__global__ __launch_bounds__(256)
void prefix_chunks(float* __restrict__ KVc, float* __restrict__ ksc) {
  const int bh = blockIdx.x, t = threadIdx.x;
  float run[32];
#pragma unroll
  for (int i = 0; i < 32; ++i) run[i] = 0.f;
  float runk = 0.f;
  for (int c = 0; c < 32; ++c) {
    float* base = KVc + ((long)bh * 32 + c) * 128 * 64;
#pragma unroll
    for (int i = 0; i < 32; ++i) {
      long idx = (long)i * 256 + t;
      float tmp = base[idx];
      base[idx] = run[i];
      run[i] += tmp;
    }
    if (t < 128) {
      float* kb = ksc + ((long)bh * 32 + c) * 128 + t;
      float tmp = *kb;
      *kb = runk;
      runk += tmp;
    }
  }
}

// ---- per-chunk output: num = Qp@KV0 + tril(Qp Kp^T)@V ; den = Qp.(ks0)+rowsum
__global__ __launch_bounds__(256)
void chunk_out(const float* __restrict__ qp, const float* __restrict__ kp,
               const float* __restrict__ vv, const float* __restrict__ KVc,
               const float* __restrict__ ksc, float* __restrict__ o) {
  __shared__ __bf16 Qs[128][132];
  __shared__ __bf16 Ks[128][132];
  __shared__ __bf16 VT[64][136];
  __shared__ __bf16 KVT[64][136];
  __shared__ __bf16 Ss[128][136];
  __shared__ float den_s[128];
  __shared__ float ks_s[128];
#if USE_ASYNC_KV
  __shared__ __align__(16) float kv_stage[128 * 64];
#endif
  const int chunk = blockIdx.x, bh = blockIdx.y;
  const int b = bh >> 3, h = bh & 7;
  const long rowBase = ((long)b * Ldim + chunk * 128) * Hdim + h;
  const float* KV0 = KVc + ((long)bh * 32 + chunk) * 128 * 64;

#if USE_ASYNC_KV
  // Kick off ASYNCcnt-tracked DMA of the 32KB KV state into LDS; it proceeds
  // while this block does the Q/K/V fills below.
  for (int idx = threadIdx.x; idx < 2048; idx += 256)
    __builtin_amdgcn_global_load_async_to_lds_b128(
        (v4i*)(uintptr_t)(KV0 + idx * 4),
        (AS3 v4i*)(uintptr_t)(&kv_stage[idx * 4]), 0, 0);
#else
  for (int idx = threadIdx.x; idx < 128 * 64; idx += 256)
    KVT[idx & 63][idx >> 6] = (__bf16)KV0[idx];
#endif
  for (int idx = threadIdx.x; idx < 128 * 128; idx += 256) {
    int l = idx >> 7, m = idx & 127;
    Qs[l][m] = (__bf16)qp[(rowBase + (long)l * Hdim) * Mdim + m];
    Ks[l][m] = (__bf16)kp[(rowBase + (long)l * Hdim) * Mdim + m];
  }
  for (int idx = threadIdx.x; idx < 128 * 64; idx += 256) {
    int l = idx >> 6, d = idx & 63;
    VT[d][l] = (__bf16)vv[(rowBase + (long)l * Hdim) * DHdim + d];
  }
  if (threadIdx.x < 128)
    ks_s[threadIdx.x] = ksc[((long)bh * 32 + chunk) * 128 + threadIdx.x];
  for (int idx = threadIdx.x; idx < 128 * 136 / 2; idx += 256)
    ((unsigned*)&Ss[0][0])[idx] = 0u;   // zero the (masked-out) upper region
#if USE_ASYNC_KV
  wait_async0();
#endif
  __syncthreads();
#if USE_ASYNC_KV
  // transpose-convert the staged f32 state to the N-major bf16 operand; the
  // den barriers below order these writes before the phase-3 reads.
  for (int idx = threadIdx.x; idx < 128 * 64; idx += 256)
    KVT[idx & 63][idx >> 6] = (__bf16)kv_stage[idx];
#endif

  const int wave = threadIdx.x >> 5, lane = threadIdx.x & 31;
  const int half = lane >> 4, r = lane & 15;
  const int ti = wave, row0 = ti * 16;
  float rden[8];
#pragma unroll
  for (int i = 0; i < 8; ++i) rden[i] = 0.f;
  for (int tj = 0; tj <= ti; ++tj) {                 // causal: only lower tiles
    v8f c = v8f_zero();
    for (int k0 = 0; k0 < 128; k0 += 32)
      c = wmma_bf16(load_a_bf16(&Qs[row0][0], 132, k0), load_b_bf16(&Ks[tj * 16][0], 132, k0), c);
#pragma unroll
    for (int v = 0; v < 8; ++v) {
      int irow = row0 + half * 8 + v;
      int jcol = tj * 16 + r;
      float val = (jcol <= irow) ? c[v] : 0.f;
      rden[v] += val;
      Ss[irow][jcol] = (__bf16)val;
    }
  }
#pragma unroll
  for (int v = 0; v < 8; ++v)
#pragma unroll
    for (int off = 1; off < 16; off <<= 1)
      rden[v] += __shfl_xor(rden[v], off, 32);
  if (r == 0)
#pragma unroll
    for (int v = 0; v < 8; ++v) den_s[row0 + half * 8 + v] = rden[v];
  __syncthreads();
  if (threadIdx.x < 128) {
    float s = 0.f;
#pragma unroll
    for (int m = 0; m < 128; ++m) s += (float)Qs[threadIdx.x][m] * ks_s[m];
    den_s[threadIdx.x] += s;
  }
  __syncthreads();

  const int nS = (ti + 2) >> 1;  // ceil((ti+1)*16 / 32) k-steps over S columns
  for (int dt = 0; dt < 4; ++dt) {
    v8f c = v8f_zero();
    for (int k0 = 0; k0 < 128; k0 += 32)
      c = wmma_bf16(load_a_bf16(&Qs[row0][0], 132, k0), load_b_bf16(&KVT[dt * 16][0], 136, k0), c);
    for (int ks = 0; ks < nS; ++ks)
      c = wmma_bf16(load_a_bf16(&Ss[row0][0], 136, ks * 32), load_b_bf16(&VT[dt * 16][0], 136, ks * 32), c);
#pragma unroll
    for (int v = 0; v < 8; ++v) {
      int lrow = row0 + half * 8 + v;
      o[(rowBase + (long)lrow * Hdim) * DHdim + dt * 16 + r] = c[v] / den_s[lrow];
    }
  }
}

extern "C" void kernel_launch(void* const* d_in, const int* in_sizes, int n_in,
                              void* d_out, int out_size, void* d_ws, size_t ws_size,
                              hipStream_t stream) {
  (void)in_sizes; (void)n_in; (void)out_size; (void)ws_size;
  const float* query = (const float*)d_in[0];
  const float* key_  = (const float*)d_in[1];
  const float* value = (const float*)d_in[2];
  const float* Wq = (const float*)d_in[3];
  const float* Wk = (const float*)d_in[4];
  const float* Wv = (const float*)d_in[5];
  const float* W0 = (const float*)d_in[6];
  const float* bq = (const float*)d_in[7];
  const float* bk = (const float*)d_in[8];
  const float* bv = (const float*)d_in[9];
  const float* b0 = (const float*)d_in[10];
  const float* proj = (const float*)d_in[11];
  float* out = (float*)d_out;

  const long NQ = (long)Bdim * Ldim * Edim;          // 8,388,608
  const long NP = (long)Bdim * Ldim * Hdim * Mdim;   // 16,777,216
  float* q   = (float*)d_ws;
  float* k   = q + NQ;
  float* v   = k + NQ;
  float* qp  = v + NQ;
  float* kp  = qp + NP;
  float* o   = kp + NP;
  float* KVc = o + NQ;                               // 32*32*128*64
  float* ksc = KVc + 8388608;                        // 32*32*128
  unsigned* bhmax = (unsigned*)(ksc + 131072);       // 32

  dim3 blk(256);
  dim3 ggemm(1024, 4);
  gemm_xwt_bias<<<ggemm, blk, 0, stream>>>(query, Wq, bq, q, Edim, Edim);
  gemm_xwt_bias<<<ggemm, blk, 0, stream>>>(key_,  Wk, bk, k, Edim, Edim);
  gemm_xwt_bias<<<ggemm, blk, 0, stream>>>(value, Wv, bv, v, Edim, Edim);
  init_bhmax<<<dim3(1), dim3(32), 0, stream>>>(bhmax);
  favor_features<<<dim3(1024), blk, 0, stream>>>(q, proj, qp, bhmax, 1);
  favor_features<<<dim3(1024), blk, 0, stream>>>(k, proj, kp, bhmax, 0);
  exp_keys<<<dim3(4096), blk, 0, stream>>>(kp, bhmax, NP);
  chunk_kv<<<dim3(32, 32), blk, 0, stream>>>(kp, v, KVc, ksc);
  prefix_chunks<<<dim3(32), blk, 0, stream>>>(KVc, ksc);
  chunk_out<<<dim3(32, 32), blk, 0, stream>>>(qp, kp, v, KVc, ksc, o);
  gemm_xwt_bias<<<ggemm, blk, 0, stream>>>(o, W0, b0, out, Edim, Edim);
}
